// TransformerProjector_45595372815105
// MI455X (gfx1250) — matrix operand order repaired and checked by reading, not compiled, not run on hardware
//
#include <hip/hip_runtime.h>
#include <hip/hip_bf16.h>

// ---------------- problem constants ----------------
#define DD   896          // hidden size
#define HH   8            // heads
#define HDP  128          // padded head dim (real 112)
#define HDR  112          // real head dim
#define LQ   1568         // M*P query tokens
#define LK   6272         // F*P kv tokens
#define FF   32           // frames
#define PP   196          // patch size
#define D4   3584         // 4*D

typedef __attribute__((ext_vector_type(16))) _Float16 v16h;
typedef __attribute__((ext_vector_type(8)))  _Float16 v8h;
typedef __attribute__((ext_vector_type(8)))  float    v8f;

// ---------------- fragment helpers (wave32 WMMA; A and B both use contiguous k = 16*(lane/16)+s) ----------------
__device__ __forceinline__ v16h ld16(const _Float16* p0, const _Float16* p1) {
  union { v16h v; v8h h[2]; } u;
  u.h[0] = *(const v8h*)p0;
  u.h[1] = *(const v8h*)p1;
  return u.v;
}

// A 16x32 f16: lane holds row M=lane%16; halves -> K = 16g+0..15
__device__ __forceinline__ v16h ld_a(const _Float16* Arow /*=A + M*lda*/, int k0, int g) {
  return ld16(Arow + k0 + 16 * g, Arow + k0 + 16 * g + 8);
}
// B 32x16 f16 (as Bt row-major [N][K]): lane holds col N=lane%16; halves -> K = 16g+0..15
__device__ __forceinline__ v16h ld_b(const _Float16* Brow /*=Bt + N*ldb*/, int k0, int g) {
  return ld16(Brow + k0 + 16 * g, Brow + k0 + 16 * g + 8);
}

__device__ __forceinline__ v8f wmma32(v16h a, v16h b, v8f c) {
  return __builtin_amdgcn_wmma_f32_16x16x32_f16(false, a, false, b, (short)0, c, false, false);
}

__device__ __forceinline__ v8f v8f_zero() {
  v8f z = {0.f, 0.f, 0.f, 0.f, 0.f, 0.f, 0.f, 0.f};
  return z;
}

__device__ __forceinline__ float redmax16(float v) {
#pragma unroll
  for (int m = 1; m < 16; m <<= 1) v = fmaxf(v, __shfl_xor(v, m, 32));
  return v;
}
__device__ __forceinline__ float redsum16(float v) {
#pragma unroll
  for (int m = 1; m < 16; m <<= 1) v += __shfl_xor(v, m, 32);
  return v;
}

// ---------------- utility kernels ----------------
__global__ void zero_h_kernel(_Float16* p, size_t n) {
  for (size_t i = (size_t)blockIdx.x * blockDim.x + threadIdx.x; i < n;
       i += (size_t)gridDim.x * blockDim.x)
    p[i] = (_Float16)0.f;
}
__global__ void zero_f_kernel(float* p, size_t n) {
  for (size_t i = (size_t)blockIdx.x * blockDim.x + threadIdx.x; i < n;
       i += (size_t)gridDim.x * blockDim.x)
    p[i] = 0.f;
}
__global__ void cvt_h_kernel(const float* __restrict__ in, _Float16* __restrict__ out, size_t n) {
  for (size_t i = (size_t)blockIdx.x * blockDim.x + threadIdx.x; i < n;
       i += (size_t)gridDim.x * blockDim.x)
    out[i] = (_Float16)in[i];
}
// in: [K][N] f32 row-major -> out: [N][K] f16 (transposed)
__global__ void trans_h_kernel(const float* __restrict__ in, _Float16* __restrict__ out,
                               int K, int N) {
  size_t total = (size_t)K * N;
  for (size_t i = (size_t)blockIdx.x * blockDim.x + threadIdx.x; i < total;
       i += (size_t)gridDim.x * blockDim.x) {
    size_t n = i / K, k = i % K;
    out[i] = (_Float16)in[k * (size_t)N + n];
  }
}

// ---------------- generic WMMA GEMM:  C = alpha*(A[M,K] x Bt[N,K]^T + bias) ----------------
// mode 0: outF[row,col] = v + resid[row,col]
// mode 1: outH[row,col] = relu(v)                       (f16)
// mode 2: outH = packed QK  [h][row][128] (h=col/112)   (f16)
// mode 3: outH = packed V^T [h][128][row]               (f16)
__global__ __launch_bounds__(128) void gemm_wmma_kernel(
    const _Float16* __restrict__ A, const _Float16* __restrict__ Bt,
    const float* __restrict__ bias, const float* __restrict__ resid,
    float* __restrict__ outF, _Float16* __restrict__ outH,
    int Mrows, int N, int K, int mode, float alpha) {
  const int wave = threadIdx.x >> 5;
  const int lane = threadIdx.x & 31;
  const int g = lane >> 4;
  const int ln = lane & 15;
  const int m0 = blockIdx.x * 16;
  const int n0 = (blockIdx.y * 4 + wave) * 16;

  const _Float16* Arow = A + (size_t)(m0 + ln) * K;
  const _Float16* Brow = Bt + (size_t)(n0 + ln) * K;

  v8f acc = v8f_zero();
  for (int k0 = 0; k0 < K; k0 += 32) {
    __builtin_prefetch(Arow + k0 + 128, 0, 1);
    __builtin_prefetch(Brow + k0 + 128, 0, 1);
    acc = wmma32(ld_a(Arow, k0, g), ld_b(Brow, k0, g), acc);
  }

#pragma unroll
  for (int r = 0; r < 8; ++r) {
    const int row = m0 + r + 8 * g;   // C/D layout: M = r + 8*(lane/16)
    const int col = n0 + ln;          //            N = lane%16
    float v = alpha * (acc[r] + bias[col]);
    if (mode == 0) {
      outF[(size_t)row * N + col] = v + resid[(size_t)row * N + col];
    } else if (mode == 1) {
      outH[(size_t)row * N + col] = (_Float16)fmaxf(v, 0.f);
    } else if (mode == 2) {
      const int h = col / HDR, dd = col % HDR;
      outH[((size_t)h * Mrows + row) * HDP + dd] = (_Float16)v;
    } else {
      const int h = col / HDR, dd = col % HDR;
      outH[((size_t)h * HDP + dd) * Mrows + row] = (_Float16)v;
    }
  }
}

// ---------------- flash attention: one wave per (head, 16-query tile) ----------------
// Qp[h][LQ][128] (pre-scaled by 1/sqrt(112)), Kp[h][LK][128], Vt[h][128][LK]
// Writes ctxh[q][h*112+dd] = softmax(QK^T) V  and ml[h*LQ+q] = {m_final, l_final}.
__global__ __launch_bounds__(32) void flash_kernel(
    const _Float16* __restrict__ Qp, const _Float16* __restrict__ Kp,
    const _Float16* __restrict__ Vt, _Float16* __restrict__ ctxh,
    float* __restrict__ ml) {
  __shared__ _Float16 lds_p[16 * 32];
  const int lane = threadIdx.x & 31;
  const int g = lane >> 4, ln = lane & 15;
  const int qt = blockIdx.x, h = blockIdx.y;

  const _Float16* Qrow = Qp + ((size_t)h * LQ + qt * 16 + ln) * HDP;
  v16h qa[4];
#pragma unroll
  for (int kk = 0; kk < 4; ++kk) qa[kk] = ld_a(Qrow, kk * 32, g);

  float mrow[8], lrow[8];
  v8f cacc[8];
#pragma unroll
  for (int r = 0; r < 8; ++r) { mrow[r] = -1e30f; lrow[r] = 0.f; cacc[r] = v8f_zero(); }

  const _Float16* Kbase = Kp + (size_t)h * LK * HDP;
  const _Float16* Vbase = Vt + (size_t)h * HDP * LK;

  for (int c = 0; c < LK / 32; ++c) {
    const int n0 = c * 32;
    const _Float16* K0 = Kbase + (size_t)(n0 + ln) * HDP;
    const _Float16* K1 = Kbase + (size_t)(n0 + 16 + ln) * HDP;
    v8f s0 = v8f_zero(), s1 = v8f_zero();
#pragma unroll
    for (int kk = 0; kk < 4; ++kk) {
      s0 = wmma32(qa[kk], ld_b(K0, kk * 32, g), s0);
      s1 = wmma32(qa[kk], ld_b(K1, kk * 32, g), s1);
    }
    // online softmax (row state replicated across the 16 lanes of each half-wave)
#pragma unroll
    for (int r = 0; r < 8; ++r) {
      float t = redmax16(fmaxf(s0[r], s1[r]));
      float mnew = fmaxf(mrow[r], t);
      float sc = __expf(mrow[r] - mnew);
      float p0 = __expf(s0[r] - mnew);
      float p1 = __expf(s1[r] - mnew);
      lrow[r] = lrow[r] * sc + redsum16(p0 + p1);
      mrow[r] = mnew;
#pragma unroll
      for (int nt = 0; nt < 8; ++nt) cacc[nt][r] *= sc;
      lds_p[(r + 8 * g) * 32 + ln] = (_Float16)p0;
      lds_p[(r + 8 * g) * 32 + 16 + ln] = (_Float16)p1;
    }
    __syncthreads();
    const v16h pa = ld_a(&lds_p[ln * 32], 0, g);  // P tile as A-fragment (16x32)
    __syncthreads();
#pragma unroll
    for (int nt = 0; nt < 8; ++nt) {
      const _Float16* Vrow = Vbase + (size_t)(nt * 16 + ln) * LK + c * 32;
      cacc[nt] = wmma32(pa, ld16(Vrow + 16 * g, Vrow + 16 * g + 8), cacc[nt]);
    }
  }

#pragma unroll
  for (int r = 0; r < 8; ++r) {
    const int q = qt * 16 + r + 8 * g;
    const float inv = 1.f / lrow[r];
#pragma unroll
    for (int nt = 0; nt < 8; ++nt) {
      const int col = nt * 16 + ln;
      if (col < HDR)
        ctxh[(size_t)q * DD + h * HDR + col] = (_Float16)(cacc[nt][r] * inv);
    }
    if (ln == 0) {
      ml[((size_t)h * LQ + q) * 2 + 0] = mrow[r];
      ml[((size_t)h * LQ + q) * 2 + 1] = lrow[r];
    }
  }
}

// ---------------- exact per-key prob column sums (for frame scores) ----------------
// One wave per (head, 16-key tile); recompute S from Qp/Kp using saved (m,l).
__global__ __launch_bounds__(32) void colsum_kernel(
    const _Float16* __restrict__ Qp, const _Float16* __restrict__ Kp,
    const float* __restrict__ ml, float* __restrict__ attn_sum) {
  const int lane = threadIdx.x & 31;
  const int g = lane >> 4, ln = lane & 15;
  const int kt = blockIdx.x, h = blockIdx.y;

  const _Float16* Krow = Kp + ((size_t)h * LK + kt * 16 + ln) * HDP;
  v16h kb[4];
#pragma unroll
  for (int kk = 0; kk < 4; ++kk) kb[kk] = ld_b(Krow, kk * 32, g);

  float csum = 0.f;
  for (int qt = 0; qt < LQ / 16; ++qt) {
    const _Float16* Qrow = Qp + ((size_t)h * LQ + qt * 16 + ln) * HDP;
    v8f s = v8f_zero();
#pragma unroll
    for (int kk = 0; kk < 4; ++kk) s = wmma32(ld_a(Qrow, kk * 32, g), kb[kk], s);
#pragma unroll
    for (int r = 0; r < 8; ++r) {
      const int q = qt * 16 + r + 8 * g;
      const float m = ml[((size_t)h * LQ + q) * 2 + 0];
      const float l = ml[((size_t)h * LQ + q) * 2 + 1];
      csum += __expf(s[r] - m) / l;
    }
  }
  csum += __shfl_xor(csum, 16, 32);  // combine row-groups of the two half-waves
  if (g == 0) atomicAdd(&attn_sum[kt * 16 + ln], csum);
}

// ---------------- layernorm over rows of length N ----------------
__global__ __launch_bounds__(256) void layernorm_kernel(
    const float* __restrict__ in, const float* __restrict__ gamma,
    const float* __restrict__ beta, float* __restrict__ outF,
    _Float16* __restrict__ outH, int N) {
  const int row = blockIdx.x;
  const float* x = in + (size_t)row * N;
  float s = 0.f, s2 = 0.f;
  for (int i = threadIdx.x; i < N; i += blockDim.x) { float v = x[i]; s += v; s2 += v * v; }
#pragma unroll
  for (int m = 1; m < 32; m <<= 1) { s += __shfl_xor(s, m, 32); s2 += __shfl_xor(s2, m, 32); }
  __shared__ float sh[18];
  const int wave = threadIdx.x >> 5, lane = threadIdx.x & 31;
  if (lane == 0) { sh[wave] = s; sh[wave + 8] = s2; }
  __syncthreads();
  if (threadIdx.x == 0) {
    float S = 0.f, S2 = 0.f;
    for (int w = 0; w < 8; ++w) { S += sh[w]; S2 += sh[w + 8]; }
    const float mu = S / N;
    sh[16] = mu;
    sh[17] = rsqrtf(fmaxf(S2 / N - mu * mu, 0.f) + 1e-12f);
  }
  __syncthreads();
  const float mu = sh[16], rinv = sh[17];
  for (int i = threadIdx.x; i < N; i += blockDim.x) {
    const float v = (x[i] - mu) * rinv * gamma[i] + beta[i];
    if (outF) outF[(size_t)row * N + i] = v;
    if (outH) outH[(size_t)row * N + i] = (_Float16)v;
  }
}

// ---------------- frame scores: mean over patch of attn_sum ----------------
__global__ __launch_bounds__(256) void frame_kernel(const float* __restrict__ attn_sum,
                                                    float* __restrict__ out) {
  const int f = blockIdx.x;
  float s = 0.f;
  for (int i = threadIdx.x; i < PP; i += blockDim.x) s += attn_sum[f * PP + i];
#pragma unroll
  for (int m = 1; m < 32; m <<= 1) s += __shfl_xor(s, m, 32);
  __shared__ float sh[8];
  const int wave = threadIdx.x >> 5, lane = threadIdx.x & 31;
  if (lane == 0) sh[wave] = s;
  __syncthreads();
  if (threadIdx.x == 0) {
    float S = 0.f;
    for (int w = 0; w < 8; ++w) S += sh[w];
    out[f] = S / (float)PP;
  }
}

// ---------------- host launch ----------------
static inline char* carve(char*& p, size_t bytes) {
  char* r = p;
  p += (bytes + 255) & ~(size_t)255;
  return r;
}

extern "C" void kernel_launch(void* const* d_in, const int* in_sizes, int n_in,
                              void* d_out, int out_size, void* d_ws, size_t ws_size,
                              hipStream_t stream) {
  (void)in_sizes; (void)n_in; (void)out_size; (void)ws_size;
  const float* img  = (const float*)d_in[0];
  const float* mem  = (const float*)d_in[1];
  const float* qW   = (const float*)d_in[2];  const float* qb = (const float*)d_in[3];
  const float* kW   = (const float*)d_in[4];  const float* kb = (const float*)d_in[5];
  const float* vW   = (const float*)d_in[6];  const float* vb = (const float*)d_in[7];
  const float* oW   = (const float*)d_in[8];  const float* ob = (const float*)d_in[9];
  const float* ln1g = (const float*)d_in[10]; const float* ln1b = (const float*)d_in[11];
  const float* mW   = (const float*)d_in[12]; const float* mb = (const float*)d_in[13];
  const float* pW   = (const float*)d_in[14]; const float* pb = (const float*)d_in[15];
  const float* ln2g = (const float*)d_in[16]; const float* ln2b = (const float*)d_in[17];
  float* outp = (float*)d_out;

  // workspace carve (~76 MB total)
  char* p = (char*)d_ws;
  _Float16* memh = (_Float16*)carve(p, (size_t)LQ * DD * 2);
  _Float16* imgh = (_Float16*)carve(p, (size_t)LK * DD * 2);   // reused later as `hidden` (same size)
  _Float16* qWt  = (_Float16*)carve(p, (size_t)DD * DD * 2);
  _Float16* kWt  = (_Float16*)carve(p, (size_t)DD * DD * 2);
  _Float16* vWt  = (_Float16*)carve(p, (size_t)DD * DD * 2);
  _Float16* oWt  = (_Float16*)carve(p, (size_t)DD * DD * 2);
  _Float16* mWt  = (_Float16*)carve(p, (size_t)DD * D4 * 2);
  _Float16* pWt  = (_Float16*)carve(p, (size_t)D4 * DD * 2);
  _Float16* Qp   = (_Float16*)carve(p, (size_t)HH * LQ * HDP * 2);
  _Float16* Kp   = (_Float16*)carve(p, (size_t)HH * LK * HDP * 2);
  _Float16* Vt   = (_Float16*)carve(p, (size_t)HH * HDP * LK * 2);
  _Float16* aoh  = (_Float16*)carve(p, (size_t)LQ * DD * 2);   // attn_out f16
  float* ml       = (float*)carve(p, (size_t)HH * LQ * 2 * 4);
  float* attn_sum = (float*)carve(p, (size_t)LK * 4);
  float* tmp1     = (float*)carve(p, (size_t)LQ * DD * 4);     // pre-LN buffer (reused for LN2 input)
  float* aoF      = (float*)carve(p, (size_t)LQ * DD * 4);     // attn_out f32
  _Float16* ctxh   = memh;  // memh dead after Q projection
  _Float16* hidden = imgh;  // imgh dead after K/V projections
  float*    tmp2   = tmp1;  // tmp1 dead after LN1

  const float qalpha = 0.09449111825230679f;  // 1/sqrt(112)

  // zero WMMA pad regions + atomic accumulator (required every call: graph replays)
  zero_h_kernel<<<2048, 256, 0, stream>>>(Qp, (size_t)HH * LQ * HDP);
  zero_h_kernel<<<2048, 256, 0, stream>>>(Kp, (size_t)HH * LK * HDP);
  zero_h_kernel<<<2048, 256, 0, stream>>>(Vt, (size_t)HH * HDP * LK);
  zero_f_kernel<<<32, 256, 0, stream>>>(attn_sum, (size_t)LK);

  // f32 -> f16 packs
  cvt_h_kernel<<<2048, 256, 0, stream>>>(mem, memh, (size_t)LQ * DD);
  cvt_h_kernel<<<4096, 256, 0, stream>>>(img, imgh, (size_t)LK * DD);
  trans_h_kernel<<<2048, 256, 0, stream>>>(qW, qWt, DD, DD);
  trans_h_kernel<<<2048, 256, 0, stream>>>(kW, kWt, DD, DD);
  trans_h_kernel<<<2048, 256, 0, stream>>>(vW, vWt, DD, DD);
  trans_h_kernel<<<2048, 256, 0, stream>>>(oW, oWt, DD, DD);
  trans_h_kernel<<<4096, 256, 0, stream>>>(mW, mWt, DD, D4);
  trans_h_kernel<<<4096, 256, 0, stream>>>(pW, pWt, D4, DD);

  // Q/K/V projections (WMMA), packed into head-major padded layouts
  gemm_wmma_kernel<<<dim3(LQ / 16, 14), 128, 0, stream>>>(memh, qWt, qb, nullptr, nullptr, Qp,
                                                          LQ, DD, DD, 2, qalpha);
  gemm_wmma_kernel<<<dim3(LK / 16, 14), 128, 0, stream>>>(imgh, kWt, kb, nullptr, nullptr, Kp,
                                                          LK, DD, DD, 2, 1.f);
  gemm_wmma_kernel<<<dim3(LK / 16, 14), 128, 0, stream>>>(imgh, vWt, vb, nullptr, nullptr, Vt,
                                                          LK, DD, DD, 3, 1.f);

  // flash attention + exact column sums for frame scores
  flash_kernel<<<dim3(LQ / 16, HH), 32, 0, stream>>>(Qp, Kp, Vt, ctxh, ml);
  colsum_kernel<<<dim3(LK / 16, HH), 32, 0, stream>>>(Qp, Kp, ml, attn_sum);

  // O projection + residual(mem) -> LN1
  gemm_wmma_kernel<<<dim3(LQ / 16, 14), 128, 0, stream>>>(ctxh, oWt, ob, mem, tmp1, nullptr,
                                                          LQ, DD, DD, 0, 1.f);
  layernorm_kernel<<<LQ, 256, 0, stream>>>(tmp1, ln1g, ln1b, aoF, aoh, DD);

  // MLP: relu(attn_out @ mW + mb) @ pW + pb + attn_out -> LN2 -> final memory
  gemm_wmma_kernel<<<dim3(LQ / 16, 56), 128, 0, stream>>>(aoh, mWt, mb, nullptr, nullptr, hidden,
                                                          LQ, D4, DD, 1, 1.f);
  gemm_wmma_kernel<<<dim3(LQ / 16, 14), 128, 0, stream>>>(hidden, pWt, pb, aoF, tmp2, nullptr,
                                                          LQ, DD, D4, 0, 1.f);
  layernorm_kernel<<<LQ, 256, 0, stream>>>(tmp2, ln2g, ln2b, outp, nullptr, DD);

  // frame scores appended after final memory
  frame_kernel<<<FF, 256, 0, stream>>>(attn_sum, outp + (size_t)LQ * DD);
}
